// LIF_18038862643638
// MI455X (gfx1250) — matrix operand order repaired and checked by reading, not compiled, not run on hardware
//
#include <hip/hip_runtime.h>
#include <math.h>

#define LEAK_F 0.01f

typedef __attribute__((ext_vector_type(16))) __bf16 v16bf;
typedef __attribute__((ext_vector_type(8)))  float  v8f;
typedef __attribute__((ext_vector_type(4)))  float  v4f;   // native vector for nontemporal stores

union FragU { uint4 q[2]; v16bf h; };

__device__ __forceinline__ unsigned int bf16_rne_bits(unsigned int u) {
    // round-to-nearest-even f32 -> bf16 bit pattern (inputs are finite normals)
    return (u + 0x7FFFu + ((u >> 16) & 1u)) >> 16;
}

// currents[T,N] = X^T[T,D] * W[D,N] - LEAK via split-bf16 WMMA (hi*hi + hi*lo + lo*hi)
// Block tile: 128(t) x 128(n), K-tile 32. 256 threads = 8 waves, each wave: 32(t) x 64(n).
__global__ __launch_bounds__(256) void lif_gemm_kernel(
    const float* __restrict__ X,   // [D, T] row-major  (X^T element (t,d) = X[d*T+t])
    const float* __restrict__ W,   // [D, N] row-major
    float* __restrict__ C,         // [T, N]
    int D, int N, int T)
{
    // LDS holds bf16 fragments in WMMA register order: per 16x(K=32) tile: lane*8 dwords
    __shared__ __align__(16) unsigned short sAhi[4096];  // 8 m-tiles x 512 halves
    __shared__ __align__(16) unsigned short sAlo[4096];
    __shared__ __align__(16) unsigned short sBhi[4096];  // 8 n-tiles x 512 halves
    __shared__ __align__(16) unsigned short sBlo[4096];

    const int tid  = threadIdx.x;
    const int n0   = blockIdx.x * 128;
    const int t0   = blockIdx.y * 128;
    const int wave = tid >> 5;
    const int lane = tid & 31;
    const int wmq  = wave & 3;   // m-tile pair: tiles {2*wmq, 2*wmq+1}
    const int wnq  = wave >> 2;  // n 64-wide half: tiles {4*wnq .. 4*wnq+3}

    v8f acc[2][4];
#pragma unroll
    for (int mi = 0; mi < 2; ++mi)
#pragma unroll
        for (int j = 0; j < 4; ++j)
            acc[mi][j] = (v8f){0.f,0.f,0.f,0.f,0.f,0.f,0.f,0.f};

    for (int k0 = 0; k0 < D; k0 += 32) {
        __syncthreads();

        // ---- stage A block: 128(t) x 32(k), coalesced along t ----
#pragma unroll
        for (int j = 0; j < 16; ++j) {
            const int i  = tid + (j << 8);
            const int tt = i & 127;
            const int kk = i >> 7;
            const float x = X[(size_t)(k0 + kk) * T + (t0 + tt)];
            const unsigned int hb = bf16_rne_bits(__float_as_uint(x));
            const float hf = __uint_as_float(hb << 16);
            const unsigned int lb = bf16_rne_bits(__float_as_uint(x - hf));
            // 16-bit A 16x32 fragment layout (ISA 7.12.2)
            const int mt = tt >> 4;
            const int ln = (tt & 15) + (((kk >> 3) & 1) << 4);
            const int vg = ((kk & 7) >> 1) + ((kk >> 4) << 2);
            const int idx = mt * 512 + ln * 16 + vg * 2 + (kk & 1);
            sAhi[idx] = (unsigned short)hb;
            sAlo[idx] = (unsigned short)lb;
        }
        // ---- stage B block: 32(k) x 128(n), coalesced along n ----
#pragma unroll
        for (int j = 0; j < 16; ++j) {
            const int i  = tid + (j << 8);
            const int nn = i & 127;
            const int kk = i >> 7;
            const float x = W[(size_t)(k0 + kk) * N + (n0 + nn)];
            const unsigned int hb = bf16_rne_bits(__float_as_uint(x));
            const float hf = __uint_as_float(hb << 16);
            const unsigned int lb = bf16_rne_bits(__float_as_uint(x - hf));
            // 16-bit B 32x16 fragment layout: lane = n + 16*((k/8)%2), vgpr = (k%8)/2 + 4*(k/16)
            const int nt = nn >> 4;
            const int ln = (nn & 15) + (((kk >> 3) & 1) << 4);
            const int vg = ((kk & 7) >> 1) + ((kk >> 4) << 2);
            const int idx = nt * 512 + ln * 16 + vg * 2 + (kk & 1);
            sBhi[idx] = (unsigned short)hb;
            sBlo[idx] = (unsigned short)lb;
        }
        __syncthreads();

        // ---- load A fragments for this wave's two m-tiles ----
        const uint4* pAh = (const uint4*)sAhi;
        const uint4* pAl = (const uint4*)sAlo;
        const uint4* pBh = (const uint4*)sBhi;
        const uint4* pBl = (const uint4*)sBlo;
        FragU aH[2], aL[2];
#pragma unroll
        for (int mi = 0; mi < 2; ++mi) {
            const int ab = (wmq * 2 + mi) * 64 + lane * 2;
            aH[mi].q[0] = pAh[ab]; aH[mi].q[1] = pAh[ab + 1];
            aL[mi].q[0] = pAl[ab]; aL[mi].q[1] = pAl[ab + 1];
        }
        // ---- sweep this wave's four n-tiles: 24 WMMAs per K-step ----
#pragma unroll
        for (int j = 0; j < 4; ++j) {
            const int bb = (wnq * 4 + j) * 64 + lane * 2;
            FragU bH, bL;
            bH.q[0] = pBh[bb]; bH.q[1] = pBh[bb + 1];
            bL.q[0] = pBl[bb]; bL.q[1] = pBl[bb + 1];
#pragma unroll
            for (int mi = 0; mi < 2; ++mi) {
                acc[mi][j] = __builtin_amdgcn_wmma_f32_16x16x32_bf16(false, aH[mi].h, false, bH.h, (short)0, acc[mi][j], false, false);
                acc[mi][j] = __builtin_amdgcn_wmma_f32_16x16x32_bf16(false, aH[mi].h, false, bL.h, (short)0, acc[mi][j], false, false);
                acc[mi][j] = __builtin_amdgcn_wmma_f32_16x16x32_bf16(false, aL[mi].h, false, bH.h, (short)0, acc[mi][j], false, false);
            }
        }
    }

    // ---- store C tiles: VGPR r -> M = r + 8*(lane/16), N = lane%16 ----
#pragma unroll
    for (int mi = 0; mi < 2; ++mi) {
#pragma unroll
        for (int j = 0; j < 4; ++j) {
            const int tBase = t0 + (wmq * 2 + mi) * 16 + ((lane >> 4) << 3);
            const int nA    = n0 + wnq * 64 + j * 16 + (lane & 15);
#pragma unroll
            for (int r = 0; r < 8; ++r) {
                C[(size_t)(tBase + r) * N + nA] = acc[mi][j][r] - LEAK_F;
            }
        }
    }
}

// sequential LIF scan over T per neuron; spikes out in [N, T] layout
__global__ __launch_bounds__(256) void lif_scan_kernel(
    const float* __restrict__ cur,  // [T, N] (already includes -LEAK)
    float* __restrict__ out,        // [N, T]
    int N, int T)
{
    const int n = blockIdx.x * blockDim.x + threadIdx.x;
    if (n >= N) return;

    float v = 0.f;
    float s[32];
    for (int tb = 0; tb < T; tb += 32) {
#pragma unroll
        for (int i = 0; i < 32; ++i) {
            const int t = tb + i;
            const float c = cur[(size_t)t * N + n];
            if (t == 0) {                 // i==0 branch: set membrane, no spike/reset
                v = c;
                s[i] = 0.f;
            } else {
                v += c;
                const bool spike = (v >= 1.0f);
                s[i] = spike ? 1.0f : 0.0f;
                if (spike || v <= 0.0f) v = 0.0f;   // reset / clamp to resting
            }
        }
        v4f* o = (v4f*)(out + (size_t)n * T + tb);
#pragma unroll
        for (int i = 0; i < 8; ++i) {
            v4f q = { s[4*i], s[4*i+1], s[4*i+2], s[4*i+3] };
            __builtin_nontemporal_store(q, o + i);
        }
    }
}

extern "C" void kernel_launch(void* const* d_in, const int* in_sizes, int n_in,
                              void* d_out, int out_size, void* d_ws, size_t ws_size,
                              hipStream_t stream) {
    (void)n_in; (void)ws_size;
    const float* X = (const float*)d_in[0];   // input_signal [D, T]
    const float* W = (const float*)d_in[1];   // weights      [D, N]
    float* out = (float*)d_out;               // spikes       [N, T]
    float* cur = (float*)d_ws;                // currents     [T, N]

    // recover D, N, T from flat sizes: i0 = D*T, i1 = D*N, o = N*T
    const double i0 = (double)in_sizes[0];
    const double i1 = (double)in_sizes[1];
    const double o  = (double)out_size;
    int N = (int)llround(sqrt(o * i1 / i0));
    if (N < 1) N = 1;
    const int T = out_size / N;
    const int D = in_sizes[1] / N;

    dim3 gemmGrid(N / 128, T / 128);
    lif_gemm_kernel<<<gemmGrid, 256, 0, stream>>>(X, W, cur, D, N, T);
    lif_scan_kernel<<<(N + 255) / 256, 256, 0, stream>>>(cur, out, N, T);
}
